// JointSentenceBiLSTM1__81114752352623
// MI455X (gfx1250) — matrix operand order repaired and checked by reading, not compiled, not run on hardware
//
#include <hip/hip_runtime.h>
#include <hip/hip_bf16.h>
#include <math.h>

// ---------------------------------------------------------------------------
// JointSentenceBiLSTM for MI455X (gfx1250, wave32, WMMA).
// B=64 L=128 E=300 H=256 EV=34 AR=36 V=30000, EVM1=33.
//
// Precision: all GEMMs in f16 (WMMA f32_16x16x32_f16), f32 accumulate.
// Sequential parts: LSTM scan = 2 persistent workgroups (one per direction,
// h kept f16 in LDS, c kept in VGPRs by static tile ownership); decode scan
// is independent per batch row -> 64 workgroups with per-b state in LDS.
// ---------------------------------------------------------------------------

typedef __attribute__((ext_vector_type(16))) _Float16 v16h;
typedef __attribute__((ext_vector_type(8)))  _Float16 v8h;
typedef __attribute__((ext_vector_type(8)))  float    v8f;

__device__ __forceinline__ v8f wmma_f16(v16h a, v16h b, v8f c) {
  // (neg_a, A, neg_b, B, c_mod, C, reuse_a, reuse_b)
  return __builtin_amdgcn_wmma_f32_16x16x32_f16(false, a, false, b, (short)0, c,
                                                false, false);
}

// A fragment: 16x32 f16, A stored row-major [M][K] with row stride `stride`.
// lane: m = lane&15, kb = (lane>>4)*16; halves 0..15 = K kb..kb+15
// (contiguous k order, matching loadB).
__device__ __forceinline__ v16h loadA(const _Float16* base, int stride,
                                      int row0, int k0, int lane) {
  int m  = lane & 15;
  int kb = (lane >> 4) << 4;
  const _Float16* p = base + (size_t)(row0 + m) * stride + k0 + kb;
  v8h lo = *(const v8h*)p;
  v8h hi = *(const v8h*)(p + 8);
  return __builtin_shufflevector(lo, hi, 0,1,2,3,4,5,6,7,8,9,10,11,12,13,14,15);
}

// B fragment: 32x16 f16 (K x N), B stored as [N][K] row-major (i.e. the
// natural layout of w_ih/w_hh/W_ar rows). lane: n = lane&15,
// kb2 = (lane>>4)*16; halves 0..15 = K kb2..kb2+15 -> one 32B load.
__device__ __forceinline__ v16h loadB(const _Float16* base, int stride,
                                      int col0, int k0, int lane) {
  int n   = lane & 15;
  int kb2 = (lane >> 4) << 4;
  return *(const v16h*)(base + (size_t)(col0 + n) * stride + k0 + kb2);
}

__device__ __forceinline__ float sigmf(float x) {
  return 1.0f / (1.0f + __expf(-x));
}

// ---- sizes -----------------------------------------------------------------
#define BB   64
#define LL   128
#define EE   300
#define EP   320     // E padded to multiple of 32
#define HH   256
#define G4   1024    // 4*H
#define H2   512     // 2*H
#define EVN  34
#define EVM  33
#define ARN  36
#define WAR  1057    // 4*H + 33 (W_ar row length)
#define WEV  545     // 2*H + 33 (W_ev row length)
#define NTOK 8192    // B*L

// ---------------------------------------------------------------------------
// K0: convert weights to padded f16.
//   wih16: [2][1024][320] (K 300..319 zero)
//   whh16: [2][1024][256]
//   wah16: [48][512]      (rows 36..47 zero) = W_ar[:, :512]
// ---------------------------------------------------------------------------
__global__ void prep_weights(const float* wih_f, const float* whh_f,
                             const float* wih_b, const float* whh_b,
                             const float* W_ar,
                             _Float16* wih16, _Float16* whh16, _Float16* wah16) {
  const int NIH = 2 * G4 * EP;
  const int NHH = 2 * G4 * HH;
  const int NAH = 48 * H2;
  int stride = gridDim.x * blockDim.x;
  for (int i = blockIdx.x * blockDim.x + threadIdx.x; i < NIH + NHH + NAH;
       i += stride) {
    if (i < NIH) {
      int d = i / (G4 * EP);
      int r = i % (G4 * EP);
      int g = r / EP, k = r % EP;
      const float* w = d ? wih_b : wih_f;
      wih16[i] = (_Float16)(k < EE ? w[g * EE + k] : 0.0f);
    } else if (i < NIH + NHH) {
      int j = i - NIH;
      int d = j / (G4 * HH);
      int r = j % (G4 * HH);
      const float* w = d ? whh_b : whh_f;
      whh16[j] = (_Float16)w[r];
    } else {
      int j = i - NIH - NHH;
      int a = j / H2, k = j % H2;
      wah16[j] = (_Float16)(a < ARN ? W_ar[(size_t)a * WAR + k] : 0.0f);
    }
  }
}

// ---------------------------------------------------------------------------
// K1: embedding gather -> padded f16 rows [token][320], token = b*L + l.
// ---------------------------------------------------------------------------
__global__ void embed_f16(const int* ids, const float* W_emb, _Float16* emb16) {
  int i = blockIdx.x * blockDim.x + threadIdx.x;
  if (i >= NTOK * EP) return;
  int tk = i / EP, k = i % EP;
  emb16[i] = (k < EE) ? (_Float16)W_emb[(size_t)ids[tk] * EE + k] : (_Float16)0.0f;
}

// ---------------------------------------------------------------------------
// K2: input projection GEMM, both directions.
//   X[d][l][b][g] = emb[token] . w_ih_d[g,:] + b_d[g]   (M=8192,N=1024,K=320)
// One 16x16 tile per wave, bias as C-init, 10 K-steps of WMMA.
// ---------------------------------------------------------------------------
__global__ __launch_bounds__(256) void gemm_xproj(const _Float16* emb16,
                                                  const _Float16* wih16,
                                                  const float* b_f,
                                                  const float* b_b,
                                                  float* xws) {
  int wave = (blockIdx.x * blockDim.x + threadIdx.x) >> 5;
  int lane = threadIdx.x & 31;
  int d   = wave >> 15;          // 32768 wave-tiles per direction
  int rem = wave & 32767;
  int mt  = rem >> 6;            // 512 M-tiles
  int nt  = rem & 63;            // 64 N-tiles
  const _Float16* Bm = wih16 + (size_t)d * G4 * EP;
  const float* bias  = d ? b_b : b_f;
  int n = lane & 15;
  int g = nt * 16 + n;
  float bg = bias[g];
  v8f acc;
#pragma unroll
  for (int r = 0; r < 8; ++r) acc[r] = bg;
#pragma unroll
  for (int ks = 0; ks < 10; ++ks) {
    v16h a = loadA(emb16, EP, mt * 16, ks * 32, lane);
    v16h b = loadB(Bm, EP, nt * 16, ks * 32, lane);
    acc = wmma_f16(a, b, acc);
  }
  int rowhalf = (lane >> 4) << 3;
#pragma unroll
  for (int r = 0; r < 8; ++r) {
    int tk = mt * 16 + rowhalf + r;
    int bb = tk >> 7, l = tk & 127;
    xws[(size_t)d * (LL * BB * G4) + ((size_t)l * BB + bb) * G4 + g] = acc[r];
  }
}

// ---------------------------------------------------------------------------
// K3: LSTM recurrence. gridDim.x = 2 (direction), 512 threads = 16 waves.
// Per step: G(64x1024) = X_t + h @ Whh^T via WMMA (h f16 in LDS), then the
// fused cell update. Each wave owns mt in 0..3, jt0..jt0+3 (j-tiles), and
// for each task computes the 4 gate tiles (i,f,g,o) so the cell update is
// purely register-local; c never leaves VGPRs.
// ---------------------------------------------------------------------------
__global__ __launch_bounds__(512) void lstm_scan(const _Float16* whh16,
                                                 const float* xws,
                                                 float* hs32, _Float16* hs16) {
  __shared__ _Float16 hbuf[BB * HH];   // 32 KB
  int d = blockIdx.x;
  const _Float16* Whh = whh16 + (size_t)d * G4 * HH;
  const float* X = xws + (size_t)d * (LL * BB * G4);
  int tid = threadIdx.x, lane = tid & 31, wave = tid >> 5;
  int mt  = wave >> 2;            // 0..3
  int jt0 = (wave & 3) * 4;       // tasks jt0..jt0+3  (jt in 0..15)
  int n = lane & 15;
  int rowhalf = (lane >> 4) << 3;

  for (int i = tid; i < BB * HH; i += 512) hbuf[i] = (_Float16)0.0f;
  float creg[4][8];
#pragma unroll
  for (int t = 0; t < 4; ++t)
#pragma unroll
    for (int r = 0; r < 8; ++r) creg[t][r] = 0.0f;

  for (int s = 0; s < LL; ++s) {
    int tr = d ? (LL - 1 - s) : s;   // time index for x and for h placement
    __syncthreads();                  // h writes from prev step visible
    v16h aF[8];
#pragma unroll
    for (int ks = 0; ks < 8; ++ks)
      aF[ks] = loadA(hbuf, HH, mt * 16, ks * 32, lane);
    __syncthreads();                  // all h reads done before new h writes
#pragma unroll
    for (int tsk = 0; tsk < 4; ++tsk) {
      int jt = jt0 + tsk;
      int j  = jt * 16 + n;
      v8f G[4];
#pragma unroll
      for (int q = 0; q < 4; ++q) {
        int gcol = q * HH + j;
        v8f c;
#pragma unroll
        for (int r = 0; r < 8; ++r) {
          int bb = mt * 16 + rowhalf + r;
          c[r] = X[((size_t)tr * BB + bb) * G4 + gcol];
        }
#pragma unroll
        for (int ks = 0; ks < 8; ++ks) {
          v16h b = loadB(Whh, HH, q * HH + jt * 16, ks * 32, lane);
          c = wmma_f16(aF[ks], b, c);
        }
        G[q] = c;
      }
#pragma unroll
      for (int r = 0; r < 8; ++r) {
        int bb = mt * 16 + rowhalf + r;
        float ig = sigmf(G[0][r]);
        float fg = sigmf(G[1][r]);
        float gg = tanhf(G[2][r]);
        float og = sigmf(G[3][r]);
        float cn = fg * creg[tsk][r] + ig * gg;
        creg[tsk][r] = cn;
        float h = og * tanhf(cn);
        hbuf[bb * HH + j] = (_Float16)h;
        size_t ho = ((size_t)bb * LL + tr) * H2 + d * HH + j;  // hs[b][t][512]
        hs32[ho] = h;
        hs16[ho] = (_Float16)h;
      }
    }
  }
}

// ---------------------------------------------------------------------------
// K4: base GEMM: base[(b,l)][a] = hs[(b,l),:512] . Wa_h[a,:] + b_ar[a]
// M=8192, N=48 (3 tiles, cols>=36 dropped), K=512 (16 WMMA steps).
// ---------------------------------------------------------------------------
__global__ __launch_bounds__(256) void gemm_base(const _Float16* hs16,
                                                 const _Float16* wah16,
                                                 const float* b_ar,
                                                 float* base) {
  int wave = (blockIdx.x * blockDim.x + threadIdx.x) >> 5;
  int lane = threadIdx.x & 31;
  int mt  = wave / 3;
  int ntB = wave % 3;
  int n = lane & 15;
  int g = ntB * 16 + n;
  float bv = (g < ARN) ? b_ar[g] : 0.0f;
  v8f acc;
#pragma unroll
  for (int r = 0; r < 8; ++r) acc[r] = bv;
#pragma unroll
  for (int ks = 0; ks < 16; ++ks) {
    v16h a = loadA(hs16, H2, mt * 16, ks * 32, lane);
    v16h b = loadB(wah16, H2, ntB * 16, ks * 32, lane);
    acc = wmma_f16(a, b, acc);
  }
  if (g < ARN) {
    int rowhalf = (lane >> 4) << 3;
#pragma unroll
    for (int r = 0; r < 8; ++r) {
      int tk = mt * 16 + rowhalf + r;
      base[(size_t)tk * ARN + g] = acc[r];
    }
  }
}

// ---------------------------------------------------------------------------
// K5: greedy decode scan. The recurrent state (g_trg, g_trg_arg) is
// independent per batch row -> one workgroup per b, all state in LDS,
// only s_barriers on the serial chain.
// ---------------------------------------------------------------------------
__global__ __launch_bounds__(256) void decode_scan(const float* hs32,
                                                   const float* base,
                                                   const float* W_ev,
                                                   const float* b_ev,
                                                   const float* W_ar,
                                                   float* out) {
  __shared__ float trig[H2];
  __shared__ float garg[LL * EVM];   // g_trg_arg[b] : 128 x 33
  __shared__ float gtrg[EVM];
  __shared__ float evl[EVN];
  __shared__ float ttl[ARN];
  __shared__ int estar;
  int b = blockIdx.x, tid = threadIdx.x;
  for (int i = tid; i < LL * EVM; i += 256) garg[i] = 0.0f;
  if (tid < EVM) gtrg[tid] = 0.0f;
  float* out_ev = out;                         // (B,L,34)
  float* out_ar = out + (size_t)BB * LL * EVN; // (B,L,L,36)

  for (int t = 0; t < LL; ++t) {
    __syncthreads();
    for (int k = tid; k < H2; k += 256)
      trig[k] = hs32[((size_t)b * LL + t) * H2 + k];
    __syncthreads();
    if (tid < EVN) {                     // ev_logit = trig.We_h^T + g.We_g^T + b
      const float* w = W_ev + (size_t)tid * WEV;
      float s = b_ev[tid];
      for (int k = 0; k < H2; ++k) s += trig[k] * w[k];
      for (int e = 0; e < EVM; ++e) s += gtrg[e] * w[H2 + e];
      evl[tid] = s;
      out_ev[((size_t)b * LL + t) * EVN + tid] = s;
    } else if (tid >= 64 && tid < 64 + ARN) {  // tt = trig . Wa_t^T
      int a = tid - 64;
      const float* w = W_ar + (size_t)a * WAR + H2;
      float s = 0.0f;
      for (int k = 0; k < H2; ++k) s += trig[k] * w[k];
      ttl[a] = s;
    }
    __syncthreads();
    if (tid == 0) {                      // argmax + g_trg update (after reads)
      int bi = 0;
      float bv = evl[0];
      for (int a = 1; a < EVN; ++a)
        if (evl[a] > bv) { bv = evl[a]; bi = a; }
      estar = bi - 1;
      if (bi >= 1) gtrg[bi - 1] = 1.0f;
    }
    __syncthreads();
    if (tid < LL) {                      // arg_logits row l = tid (owned)
      int l = tid;
      float loc[ARN];
      for (int a = 0; a < ARN; ++a) {
        const float* wg = W_ar + (size_t)a * WAR + G4;
        float s = base[((size_t)b * LL + l) * ARN + a] + ttl[a];
        for (int e = 0; e < EVM; ++e) s += garg[l * EVM + e] * wg[e];
        loc[a] = s;
      }
      int bi = 0;
      float bv = loc[0];
      for (int a = 1; a < ARN; ++a)
        if (loc[a] > bv) { bv = loc[a]; bi = a; }
      size_t o = (((size_t)b * LL + t) * LL + l) * ARN;
      for (int a = 0; a < ARN; ++a) out_ar[o + a] = loc[a];
      if (bi > 0 && estar >= 0) garg[l * EVM + estar] = 1.0f;  // own row only
    }
  }
}

// ---------------------------------------------------------------------------
extern "C" void kernel_launch(void* const* d_in, const int* in_sizes, int n_in,
                              void* d_out, int out_size, void* d_ws,
                              size_t ws_size, hipStream_t stream) {
  (void)in_sizes; (void)n_in; (void)out_size; (void)ws_size;
  const int*   ids   = (const int*)  d_in[0];
  const float* W_emb = (const float*)d_in[1];
  const float* wih_f = (const float*)d_in[2];
  const float* whh_f = (const float*)d_in[3];
  const float* b_f   = (const float*)d_in[4];
  const float* wih_b = (const float*)d_in[5];
  const float* whh_b = (const float*)d_in[6];
  const float* b_b   = (const float*)d_in[7];
  const float* W_ev  = (const float*)d_in[8];
  const float* b_ev  = (const float*)d_in[9];
  const float* W_ar  = (const float*)d_in[10];
  const float* b_ar  = (const float*)d_in[11];
  float* out = (float*)d_out;

  char* ws = (char*)d_ws;
  size_t off = 0;
  auto carve = [&](size_t bytes) -> char* {
    char* p = ws + off;
    off += (bytes + 255) & ~(size_t)255;
    return p;
  };
  _Float16* wih16 = (_Float16*)carve((size_t)2 * G4 * EP * 2);   // 1.25 MB
  _Float16* whh16 = (_Float16*)carve((size_t)2 * G4 * HH * 2);   // 1.0 MB
  _Float16* wah16 = (_Float16*)carve((size_t)48 * H2 * 2);       // 48 KB
  _Float16* emb16 = (_Float16*)carve((size_t)NTOK * EP * 2);     // 5 MB
  float*    xws   = (float*)   carve((size_t)2 * LL * BB * G4 * 4); // 64 MB
  float*    hs32  = (float*)   carve((size_t)NTOK * H2 * 4);     // 16 MB
  _Float16* hs16  = (_Float16*)carve((size_t)NTOK * H2 * 2);     // 8 MB
  float*    baseb = (float*)   carve((size_t)NTOK * ARN * 4);    // 1.1 MB

  // K0: weight conversion (grid-stride)
  prep_weights<<<1024, 256, 0, stream>>>(wih_f, whh_f, wih_b, whh_b, W_ar,
                                         wih16, whh16, wah16);
  // K1: embedding gather (8192*320 elements)
  embed_f16<<<(NTOK * EP + 255) / 256, 256, 0, stream>>>(ids, W_emb, emb16);
  // K2: input projection, 65536 wave-tiles / 8 waves per block
  gemm_xproj<<<8192, 256, 0, stream>>>(emb16, wih16, b_f, b_b, xws);
  // K3: BiLSTM recurrence, one block per direction
  lstm_scan<<<2, 512, 0, stream>>>(whh16, xws, hs32, hs16);
  // K4: base GEMM, 1536 wave-tiles / 8 waves per block
  gemm_base<<<192, 256, 0, stream>>>(hs16, wah16, b_ar, baseb);
  // K5: decode scan, one block per batch row
  decode_scan<<<BB, 256, 0, stream>>>(hs32, baseb, W_ev, b_ev, W_ar, out);
}
